// SimilarityModel_5909874999428
// MI455X (gfx1250) — matrix operand order repaired and checked by reading, not compiled, not run on hardware
//
#include <hip/hip_runtime.h>
#include <hip/hip_bf16.h>

typedef _Float16 v16h __attribute__((ext_vector_type(16)));
typedef _Float16 v8h  __attribute__((ext_vector_type(8)));
typedef float    v8f  __attribute__((ext_vector_type(8)));

#define BB 4
#define HH 256
#define WW 512
#define HWTOT (HH * WW)

// ---------------------------------------------------------------------------
// Weight preparation: (OC,IC,3,3) f32 -> [tap][oc][ic] fp16  (B matrix, col-major)
// ---------------------------------------------------------------------------
__global__ void k_prep_wconv(const float* __restrict__ w, _Float16* __restrict__ o) {
  int i = blockIdx.x * blockDim.x + threadIdx.x;
  if (i >= 9 * 64 * 64) return;
  int k9 = i >> 12;                // tap = kh*3+kw
  int r  = i & 4095;
  int oc = r >> 6, ic = r & 63;
  int kh = k9 / 3, kw = k9 % 3;
  o[i] = (_Float16)w[((oc * 64 + ic) * 3 + kh) * 3 + kw];
}

// w1d (E=64, C=64, F=256) f32 -> wp[e][f*64+c] fp16
__global__ void k_prep_wproj(const float* __restrict__ w1d, _Float16* __restrict__ o) {
  int i = blockIdx.x * blockDim.x + threadIdx.x;
  if (i >= 64 * 16384) return;
  int e = i >> 14;
  int k = i & 16383;
  int f = k >> 6, c = k & 63;
  o[i] = (_Float16)w1d[(e * 64 + c) * 256 + f];
}

// ---------------------------------------------------------------------------
// GroupNorm: stats (sum, sumsq) per (b, group), NHWC input
// ---------------------------------------------------------------------------
__global__ __launch_bounds__(256) void k_gn_stats(const float* __restrict__ x,
                                                  float* __restrict__ stats,
                                                  int C, int G) {
  const int bg = blockIdx.y;
  const int b = bg / G, g = bg % G;
  const int cg = C / G;
  const long n = (long)HWTOT * cg;
  float s = 0.f, s2 = 0.f;
  for (long i = (long)blockIdx.x * blockDim.x + threadIdx.x; i < n;
       i += (long)gridDim.x * blockDim.x) {
    long hw = i / cg;
    int ci = (int)(i % cg);
    float v = x[((long)b * HWTOT + hw) * C + g * cg + ci];
    s += v;
    s2 += v * v;
  }
  __shared__ float rs[256], rq[256];
  rs[threadIdx.x] = s;
  rq[threadIdx.x] = s2;
  __syncthreads();
  for (int st = 128; st > 0; st >>= 1) {
    if (threadIdx.x < st) {
      rs[threadIdx.x] += rs[threadIdx.x + st];
      rq[threadIdx.x] += rq[threadIdx.x + st];
    }
    __syncthreads();
  }
  if (threadIdx.x == 0) {
    atomicAdd(&stats[2 * bg + 0], rs[0]);
    atomicAdd(&stats[2 * bg + 1], rq[0]);
  }
}

// normalize + affine + ReLU + fp16 pack (produces WMMA-ready NHWC fp16)
__global__ __launch_bounds__(256) void k_gn_apply(const float* __restrict__ x,
                                                  const float* __restrict__ stats,
                                                  const float* __restrict__ scale,
                                                  const float* __restrict__ bias,
                                                  _Float16* __restrict__ out,
                                                  int C, int G, float invCnt) {
  int idx = blockIdx.x * blockDim.x + threadIdx.x;
  int total = BB * HWTOT * C;
  if (idx >= total) return;
  int c = idx % C;
  int b = idx / (C * HWTOT);
  int g = c / (C / G);
  int bg = b * G + g;
  float mu = stats[2 * bg + 0] * invCnt;
  float var = stats[2 * bg + 1] * invCnt - mu * mu;
  float r = rsqrtf(var + 1e-5f);
  float v = (x[idx] - mu) * r * scale[c] + bias[c];
  out[idx] = (_Float16)fmaxf(v, 0.f);
}

__global__ void k_f32_to_f16(const float* __restrict__ x, _Float16* __restrict__ o, int n) {
  int i = blockIdx.x * blockDim.x + threadIdx.x;
  if (i < n) o[i] = (_Float16)x[i];
}

// ---------------------------------------------------------------------------
// RCB0 conv1: 1 -> 64 channels (K=9, too small for WMMA; direct VALU conv)
// ---------------------------------------------------------------------------
__global__ __launch_bounds__(256) void k_conv_c1(const _Float16* __restrict__ a,  // [B][H][W]
                                                 const float* __restrict__ w,     // [64][1][3][3]
                                                 const float* __restrict__ bias,
                                                 float* __restrict__ out)         // [B][H][W][64]
{
  __shared__ float wsh[576];
  __shared__ float bsh[64];
  for (int i = threadIdx.x; i < 576; i += 256) wsh[i] = w[i];
  if (threadIdx.x < 64) bsh[threadIdx.x] = bias[threadIdx.x];
  __syncthreads();
  int gi = blockIdx.x * 256 + threadIdx.x;
  if (gi >= BB * HWTOT) return;
  int b = gi / HWTOT, hw = gi % HWTOT;
  int h = hw / WW, wc = hw % WW;
  float in9[9];
#pragma unroll
  for (int kh = 0; kh < 3; ++kh)
#pragma unroll
    for (int kw = 0; kw < 3; ++kw) {
      int gh = h + kh - 1, gw = wc + kw - 1;
      in9[kh * 3 + kw] = (gh >= 0 && gh < HH && gw >= 0 && gw < WW)
                             ? (float)a[(b * HH + gh) * WW + gw] : 0.f;
    }
  float* o = &out[(size_t)gi * 64];
  for (int oc = 0; oc < 64; ++oc) {
    float s = bsh[oc];
#pragma unroll
    for (int j = 0; j < 9; ++j) s += in9[j] * wsh[oc * 9 + j];
    o[oc] = s;
  }
}

// ---------------------------------------------------------------------------
// 3x3 conv, 64->64 channels, implicit GEMM with v_wmma_f32_16x16x32_f16.
// Block: 16x16 pixel tile x 64 oc, 8 waves (2 output rows each).
// K-loop: 9 taps x (2 x 32 ic). LDS: 18x18x64 halo patch (stride 72) +
// double-buffered [oc][ic] weight slice. 63.6 KB LDS total.
// ---------------------------------------------------------------------------
__global__ __launch_bounds__(256) void k_conv3x3_wmma(
    const _Float16* __restrict__ act,   // [B][H][W][64] fp16
    const _Float16* __restrict__ wt,    // [9][64 oc][64 ic] fp16
    const float* __restrict__ bias,     // [64]
    const float* __restrict__ res,      // residual: NHWC64 (mode 1) or [B][H][W] (mode 2)
    int resMode,                        // 0 none, 1 full, 2 broadcast C=1
    float* __restrict__ out)            // [B][H][W][64]
{
  __shared__ __align__(16) _Float16 patch[18 * 18 * 72];
  __shared__ __align__(16) _Float16 wsl[2][64 * 72];
  const int b = blockIdx.z;
  const int h0 = blockIdx.y * 16;
  const int w0 = blockIdx.x * 16;
  const int tid = threadIdx.x;
  const int lane = tid & 31, wid = tid >> 5;
  const int mrow = lane & 15;    // A/C M and B N lane index
  const int khalf = lane >> 4;   // which K/M half this lane owns

  // stage 18x18x64 halo patch, zero-padded
  for (int p = tid; p < 18 * 18; p += 256) {
    const int ph = p / 18, pw = p % 18;
    const int gh = h0 + ph - 1, gw = w0 + pw - 1;
    v8h* dst = (v8h*)&patch[p * 72];
    if (gh >= 0 && gh < HH && gw >= 0 && gw < WW) {
      const v8h* src = (const v8h*)&act[((b * HH + gh) * WW + gw) * 64];
#pragma unroll
      for (int q = 0; q < 8; ++q) dst[q] = src[q];
    } else {
      v8h z;
#pragma unroll
      for (int q = 0; q < 8; ++q) z[q] = (_Float16)0.f;
#pragma unroll
      for (int q = 0; q < 8; ++q) dst[q] = z;
    }
  }

  v8f acc[2][4];
#pragma unroll
  for (int mt = 0; mt < 2; ++mt)
#pragma unroll
    for (int nt = 0; nt < 4; ++nt)
#pragma unroll
      for (int q = 0; q < 8; ++q) acc[mt][nt][q] = 0.f;

  for (int t9 = 0; t9 < 9; ++t9) {
    {  // stage [oc][ic] weight slice for this tap (double-buffered)
      const _Float16* s = &wt[t9 * 4096];
      _Float16* d = wsl[t9 & 1];
      const int e = tid >> 2;
      const int ch = (tid & 3) * 16;
      const v8h* s8 = (const v8h*)&s[e * 64 + ch];
      v8h* d8 = (v8h*)&d[e * 72 + ch];
      d8[0] = s8[0];
      d8[1] = s8[1];
    }
    __syncthreads();

    const int kh = t9 / 3, kw = t9 % 3;
    // A fragments: lane m=mrow, K chunks (khalf*8) and (16+khalf*8) per kstep
    v16h afr[2][2];
#pragma unroll
    for (int mt = 0; mt < 2; ++mt) {
      const _Float16* pr = &patch[((wid * 2 + mt + kh) * 18 + (mrow + kw)) * 72];
#pragma unroll
      for (int ks = 0; ks < 2; ++ks) {
        union { v16h v; v8h h[2]; } u;
        u.h[0] = *(const v8h*)&pr[ks * 32 + khalf * 8];
        u.h[1] = *(const v8h*)&pr[ks * 32 + 16 + khalf * 8];
        afr[mt][ks] = u.v;
      }
    }
    const _Float16* wr = wsl[t9 & 1];
#pragma unroll
    for (int ks = 0; ks < 2; ++ks) {
#pragma unroll
      for (int nt = 0; nt < 4; ++nt) {
        // B fragment: lane N = nt*16+mrow, K chunks ks*32 + (khalf*8) and (16+khalf*8)
        union { v16h v; v8h h[2]; } u;
        const _Float16* bp = &wr[(nt * 16 + mrow) * 72 + ks * 32 + khalf * 8];
        u.h[0] = *(const v8h*)bp;
        u.h[1] = *(const v8h*)(bp + 16);
#pragma unroll
        for (int mt = 0; mt < 2; ++mt)
          acc[mt][nt] = __builtin_amdgcn_wmma_f32_16x16x32_f16(
              false, afr[mt][ks], false, u.v, (short)0, acc[mt][nt], false, false);
      }
    }
  }

  // epilogue: bias + optional residual, f32 NHWC store
#pragma unroll
  for (int mt = 0; mt < 2; ++mt) {
    const int h = h0 + wid * 2 + mt;
#pragma unroll
    for (int nt = 0; nt < 4; ++nt) {
      const int oc = nt * 16 + mrow;
      const float bv = bias[oc];
#pragma unroll
      for (int r = 0; r < 8; ++r) {
        const int w = w0 + r + khalf * 8;        // C layout: M = r + (lane>=16)*8
        const int pix = (b * HH + h) * WW + w;
        float v = acc[mt][nt][r] + bv;
        if (resMode == 1)      v += res[(size_t)pix * 64 + oc];
        else if (resMode == 2) v += res[pix];
        out[(size_t)pix * 64 + oc] = v;
      }
    }
  }
}

// ---------------------------------------------------------------------------
// Projection einsum 'btcf,ecf->bte' as WMMA GEMM: M=t(128/block), N=e(64),
// K = f*64+c (16384), f-sliced with double-buffered LDS staging.
// ---------------------------------------------------------------------------
__global__ __launch_bounds__(256) void k_proj_wmma(
    const _Float16* __restrict__ act,  // [B][256 f][512 t][64 c] fp16
    const _Float16* __restrict__ wp,   // [64 e][16384 k] fp16
    const float* __restrict__ b1d,     // [64]
    float* __restrict__ y)             // [B][512 t][64 e]
{
  __shared__ __align__(16) _Float16 xs[2][128 * 72];
  __shared__ __align__(16) _Float16 wsh[2][64 * 72];
  const int b = blockIdx.x >> 2;
  const int t0 = (blockIdx.x & 3) * 128;
  const int tid = threadIdx.x, lane = tid & 31, wid = tid >> 5;
  const int mrow = lane & 15, khalf = lane >> 4;

  v8f acc[4];
#pragma unroll
  for (int nt = 0; nt < 4; ++nt)
#pragma unroll
    for (int q = 0; q < 8; ++q) acc[nt][q] = 0.f;

  for (int f = 0; f < 256; ++f) {
    {  // stage 128 t x 64 c activation slice
      const int row = tid >> 1, ch = (tid & 1) * 32;
      const v8h* s = (const v8h*)&act[(((b * 256 + f) * 512) + (t0 + row)) * 64 + ch];
      v8h* d = (v8h*)&xs[f & 1][row * 72 + ch];
      d[0] = s[0]; d[1] = s[1]; d[2] = s[2]; d[3] = s[3];
    }
    {  // stage 64 e x 64 c weight slice
      const int e = tid >> 2, ch = (tid & 3) * 16;
      const v8h* s = (const v8h*)&wp[e * 16384 + f * 64 + ch];
      v8h* d = (v8h*)&wsh[f & 1][e * 72 + ch];
      d[0] = s[0]; d[1] = s[1];
    }
    __syncthreads();
    const _Float16* xr = &xs[f & 1][(wid * 16 + mrow) * 72];
#pragma unroll
    for (int ks = 0; ks < 2; ++ks) {
      union { v16h v; v8h h[2]; } ua;
      ua.h[0] = *(const v8h*)&xr[ks * 32 + khalf * 8];
      ua.h[1] = *(const v8h*)&xr[ks * 32 + 16 + khalf * 8];
#pragma unroll
      for (int nt = 0; nt < 4; ++nt) {
        union { v16h v; v8h h[2]; } ub;
        const _Float16* bp = &wsh[f & 1][(nt * 16 + mrow) * 72 + ks * 32 + khalf * 8];
        ub.h[0] = *(const v8h*)bp;
        ub.h[1] = *(const v8h*)(bp + 16);
        acc[nt] = __builtin_amdgcn_wmma_f32_16x16x32_f16(
            false, ua.v, false, ub.v, (short)0, acc[nt], false, false);
      }
    }
  }
#pragma unroll
  for (int nt = 0; nt < 4; ++nt) {
    const int e = nt * 16 + mrow;
    const float bv = b1d[e];
#pragma unroll
    for (int r = 0; r < 8; ++r) {
      const int t = t0 + wid * 16 + r + khalf * 8;
      y[((b * 512) + t) * 64 + e] = acc[nt][r] + bv;
    }
  }
}

// L2-normalize audio embedding over e per (b,t)
__global__ void k_anorm(const float* __restrict__ y, float* __restrict__ an) {
  int i = blockIdx.x * blockDim.x + threadIdx.x;
  if (i >= BB * 512) return;
  const float* p = &y[i * 64];
  float s = 0.f;
#pragma unroll
  for (int e = 0; e < 64; ++e) s += p[e] * p[e];
  float inv = 1.f / fmaxf(sqrtf(s), 1e-12f);
  float* o = &an[i * 64];
#pragma unroll
  for (int e = 0; e < 64; ++e) o[e] = p[e] * inv;
}

// ---------------------------------------------------------------------------
// Text encoder (FLOP-negligible, plain VALU)
// ---------------------------------------------------------------------------
__global__ void k_text_embed(const int* __restrict__ tok, const float* __restrict__ emb,
                             float* __restrict__ h0, int R) {
  int i = blockIdx.x * blockDim.x + threadIdx.x;
  if (i >= R * 192) return;
  int r = i / 192, j = i % 192;
  int t = tok[r * 3 + j / 64];
  h0[i] = emb[t * 64 + (j & 63)];
}

__global__ void k_text_mm_relu(const float* __restrict__ h0, const float* __restrict__ w1,
                               const float* __restrict__ b1, float* __restrict__ h1, int R) {
  int i = blockIdx.x * blockDim.x + threadIdx.x;
  if (i >= R * 192) return;
  int r = i / 192, j = i % 192;
  float s = b1[j];
  const float* hr = &h0[r * 192];
  for (int k = 0; k < 192; ++k) s += hr[k] * w1[k * 192 + j];
  h1[i] = fmaxf(s, 0.f);
}

__global__ void k_text_mm2(const float* __restrict__ h1, const float* __restrict__ w2,
                           const float* __restrict__ b2, float* __restrict__ h2, int R) {
  int i = blockIdx.x * blockDim.x + threadIdx.x;
  if (i >= R * 64) return;
  int r = i / 64, e = i % 64;
  float s = b2[e];
  const float* hr = &h1[r * 192];
  for (int k = 0; k < 192; ++k) s += hr[k] * w2[k * 64 + e];
  h2[i] = s;
}

__global__ void k_text_norm(const float* __restrict__ h2, float* __restrict__ o, int R) {
  int r = blockIdx.x * blockDim.x + threadIdx.x;
  if (r >= R) return;
  const float* p = &h2[r * 64];
  float s = 0.f;
#pragma unroll
  for (int e = 0; e < 64; ++e) s += p[e] * p[e];
  float inv = 1.f / fmaxf(sqrtf(s), 1e-12f);
  float* q = &o[r * 64];
#pragma unroll
  for (int e = 0; e < 64; ++e) q[e] = p[e] * inv;
}

// ---------------------------------------------------------------------------
// Final similarity products
// ---------------------------------------------------------------------------
__global__ void k_pa(const float* __restrict__ P, const float* __restrict__ an,
                     const int* __restrict__ len_pos, float* __restrict__ o) {
  int i = blockIdx.x * blockDim.x + threadIdx.x;
  if (i >= 400 * 512) return;
  int p = i >> 9, t = i & 511;
  int bsel = BB - 1, cum = 0;
  for (int b = 0; b < BB; ++b) {
    cum += len_pos[b];
    if (p < cum) { bsel = b; break; }
  }
  const float* ap = &an[(bsel * 512 + t) * 64];
  const float* pp = &P[p * 64];
  float s = 0.f;
#pragma unroll
  for (int e = 0; e < 64; ++e) s += pp[e] * ap[e];
  o[i] = s;
}

__global__ void k_na(const float* __restrict__ N, const float* __restrict__ an,
                     float* __restrict__ o) {
  int i = blockIdx.x * blockDim.x + threadIdx.x;
  if (i >= 4000 * 512) return;
  int r = i >> 9, t = i & 511;
  int b = r / 1000;
  const float* ap = &an[(b * 512 + t) * 64];
  const float* np = &N[r * 64];
  float s = 0.f;
#pragma unroll
  for (int e = 0; e < 64; ++e) s += np[e] * ap[e];
  o[i] = s;
}

// ---------------------------------------------------------------------------
// Host orchestration
// ---------------------------------------------------------------------------
extern "C" void kernel_launch(void* const* d_in, const int* in_sizes, int n_in,
                              void* d_out, int out_size, void* d_ws, size_t ws_size,
                              hipStream_t stream) {
  (void)in_sizes; (void)n_in; (void)out_size; (void)ws_size;
  // pytree-flattened input order (top-level insertion, params leaves key-sorted)
  const float* spec    = (const float*)d_in[0];
  const int*   pos     = (const int*)d_in[1];
  const int*   len_pos = (const int*)d_in[2];
  const int*   neg     = (const int*)d_in[3];
  const float* b1d     = (const float*)d_in[4];
  const float* w1d     = (const float*)d_in[45];
  const float* tb1     = (const float*)d_in[46];
  const float* tb2     = (const float*)d_in[47];
  const float* temb    = (const float*)d_in[48];
  const float* tw1     = (const float*)d_in[49];
  const float* tw2     = (const float*)d_in[50];

  char* base = (char*)d_ws;
  size_t off = 0;
  auto carve = [&](size_t bytes) -> char* {
    char* p = base + off;
    off += (bytes + 255) & ~(size_t)255;
    return p;
  };
  const size_t NACT = (size_t)BB * HWTOT * 64;
  float*    X   = (float*)carve(NACT * 4);            // residual / layer state, NHWC f32
  float*    Hb  = (float*)carve(NACT * 4);            // conv1 output, NHWC f32
  _Float16* ACT = (_Float16*)carve(NACT * 2);         // WMMA-ready fp16 NHWC
  float*    ST  = (float*)carve(10 * 32 * 4);         // GN stats strips
  _Float16* WC  = (_Float16*)carve((size_t)9 * 36864 * 2);
  _Float16* WP  = (_Float16*)carve((size_t)64 * 16384 * 2);
  float*    Y   = (float*)carve((size_t)BB * 512 * 64 * 4);
  float*    AN  = (float*)carve((size_t)BB * 512 * 64 * 4);
  float*    T0  = (float*)carve((size_t)4000 * 192 * 4);
  float*    T1  = (float*)carve((size_t)4000 * 192 * 4);
  float*    PB  = (float*)carve((size_t)400 * 64 * 4);
  float*    NB  = (float*)carve((size_t)4000 * 64 * 4);

  hipMemsetAsync(ST, 0, 10 * 32 * sizeof(float), stream);

  // --- weight prep (9 WMMA convs + projection) ---
  k_prep_wconv<<<144, 256, 0, stream>>>((const float*)d_in[8], WC);  // rcb0 conv2
  for (int i = 1; i < 5; ++i) {
    int bi = 5 + 8 * i;
    k_prep_wconv<<<144, 256, 0, stream>>>((const float*)d_in[bi + 1], WC + (2 * i - 1) * 36864);
    k_prep_wconv<<<144, 256, 0, stream>>>((const float*)d_in[bi + 3], WC + (2 * i) * 36864);
  }
  k_prep_wproj<<<4096, 256, 0, stream>>>(w1d, WP);

  const dim3 convGrid(WW / 16, HH / 16, BB);
  const int elemBlocks = (BB * HWTOT * 64 + 255) / 256;

  // --- RCB 0 (cin=1) ---
  k_gn_stats<<<dim3(128, BB * 1), 256, 0, stream>>>(spec, ST + 0 * 32, 1, 1);
  k_gn_apply<<<(BB * HWTOT + 255) / 256, 256, 0, stream>>>(
      spec, ST + 0 * 32, (const float*)d_in[10], (const float*)d_in[9], ACT, 1, 1,
      1.f / (float)HWTOT);
  k_conv_c1<<<BB * HWTOT / 256, 256, 0, stream>>>(ACT, (const float*)d_in[6],
                                                  (const float*)d_in[5], Hb);
  k_gn_stats<<<dim3(128, BB * 4), 256, 0, stream>>>(Hb, ST + 1 * 32, 64, 4);
  k_gn_apply<<<elemBlocks, 256, 0, stream>>>(Hb, ST + 1 * 32, (const float*)d_in[12],
                                             (const float*)d_in[11], ACT, 64, 4,
                                             1.f / (float)(HWTOT * 16));
  k_conv3x3_wmma<<<convGrid, 256, 0, stream>>>(ACT, WC, (const float*)d_in[7], spec, 2, X);

  // --- RCBs 1..4 ---
  for (int i = 1; i < 5; ++i) {
    int bi = 5 + 8 * i;
    k_gn_stats<<<dim3(128, BB * 4), 256, 0, stream>>>(X, ST + (2 * i) * 32, 64, 4);
    k_gn_apply<<<elemBlocks, 256, 0, stream>>>(X, ST + (2 * i) * 32,
                                               (const float*)d_in[bi + 5],
                                               (const float*)d_in[bi + 4], ACT, 64, 4,
                                               1.f / (float)(HWTOT * 16));
    k_conv3x3_wmma<<<convGrid, 256, 0, stream>>>(ACT, WC + (2 * i - 1) * 36864,
                                                 (const float*)d_in[bi + 0], nullptr, 0, Hb);
    k_gn_stats<<<dim3(128, BB * 4), 256, 0, stream>>>(Hb, ST + (2 * i + 1) * 32, 64, 4);
    k_gn_apply<<<elemBlocks, 256, 0, stream>>>(Hb, ST + (2 * i + 1) * 32,
                                               (const float*)d_in[bi + 7],
                                               (const float*)d_in[bi + 6], ACT, 64, 4,
                                               1.f / (float)(HWTOT * 16));
    k_conv3x3_wmma<<<convGrid, 256, 0, stream>>>(ACT, WC + (2 * i) * 36864,
                                                 (const float*)d_in[bi + 2], X, 1, X);
  }

  // --- projection + audio L2 norm ---
  k_f32_to_f16<<<elemBlocks, 256, 0, stream>>>(X, ACT, BB * HWTOT * 64);
  k_proj_wmma<<<BB * 4, 256, 0, stream>>>(ACT, WP, b1d, Y);
  k_anorm<<<(BB * 512 + 255) / 256, 256, 0, stream>>>(Y, AN);

  // --- text encoder: positives (400 rows) ---
  k_text_embed<<<(400 * 192 + 255) / 256, 256, 0, stream>>>(pos, temb, T0, 400);
  k_text_mm_relu<<<(400 * 192 + 255) / 256, 256, 0, stream>>>(T0, tw1, tb1, T1, 400);
  k_text_mm2<<<(400 * 64 + 255) / 256, 256, 0, stream>>>(T1, tw2, tb2, T0, 400);
  k_text_norm<<<(400 + 255) / 256, 256, 0, stream>>>(T0, PB, 400);
  // --- text encoder: negatives (4000 rows) ---
  k_text_embed<<<(4000 * 192 + 255) / 256, 256, 0, stream>>>(neg, temb, T0, 4000);
  k_text_mm_relu<<<(4000 * 192 + 255) / 256, 256, 0, stream>>>(T0, tw1, tb1, T1, 4000);
  k_text_mm2<<<(4000 * 64 + 255) / 256, 256, 0, stream>>>(T1, tw2, tb2, T0, 4000);
  k_text_norm<<<(4000 + 255) / 256, 256, 0, stream>>>(T0, NB, 4000);

  // --- final products: PA (400x512) then NA (4000x512) ---
  float* out = (float*)d_out;
  k_pa<<<(400 * 512 + 255) / 256, 256, 0, stream>>>(PB, AN, len_pos, out);
  k_na<<<(4000 * 512 + 255) / 256, 256, 0, stream>>>(NB, AN, out + 400 * 512);
}
